// DecoderWithRelativePositionalAttentionLayer_56762287784054
// MI455X (gfx1250) — matrix operand order repaired and checked by reading, not compiled, not run on hardware
//
#include <hip/hip_runtime.h>
#include <cstddef>
#include <cstdint>

typedef __bf16 bf16;
typedef __attribute__((ext_vector_type(16))) __bf16 v16bf;
typedef __attribute__((ext_vector_type(8)))  __bf16 v8bf;
typedef __attribute__((ext_vector_type(8)))  float  v8f;
typedef __attribute__((ext_vector_type(4)))  int    v4i;

namespace {
constexpr int NB   = 8;            // batch
constexpr int LQ   = 384;
constexpr int LK   = 384;
constexpr int D    = 512;
constexpr int H    = 8;
constexpr int DIM  = 64;
constexpr int HID  = 2048;
constexpr int REL  = 100;
constexpr int NREL = 2 * REL + 1;  // 201
constexpr int ROWS = NB * LQ;      // 3072
constexpr float SCALE = 0.125f;    // 1/sqrt(64)
constexpr float NEGV  = -1.0e9f;
constexpr int BSTRIDE = 40;        // padded LDS row stride (bf16) -> 80B, 16B-aligned
}

// ---------------------------------------------------------------------------
// CDNA5 async global->LDS staging (GLOBAL_LOAD_ASYNC_TO_LDS_B128, ASYNCcnt).
// Guarded by __has_builtin: falls back to register staging if unavailable.
// ---------------------------------------------------------------------------
#if __has_builtin(__builtin_amdgcn_global_load_async_to_lds_b128) && \
    __has_builtin(__builtin_amdgcn_s_wait_asynccnt)
#define USE_ASYNC_LDS 1
#else
#define USE_ASYNC_LDS 0
#endif

#if USE_ASYNC_LDS
typedef __attribute__((address_space(1))) v4i gv4i;
typedef __attribute__((address_space(3))) v4i lv4i;
__device__ __forceinline__ void stage16_async(const bf16* g, bf16* l) {
  __builtin_amdgcn_global_load_async_to_lds_b128((gv4i*)g, (lv4i*)l, 0, 0);
}
__device__ __forceinline__ void wait_async0() {
  __builtin_amdgcn_s_wait_asynccnt(0);
}
#endif

// ---------------------------------------------------------------------------
// WMMA fragment helpers (CDNA5 v_wmma_f32_16x16x32_bf16, wave32)
// A frag (16x32 MxK): lane -> row M=lane&15,
//                     K = k0 + (lane>=16?16:0) + {0..15} (contiguous).
// B frag (32x16 KxN) from B^T[N,K] rows: lane -> col N=lane&15,
//                     K = k0 + (lane>=16?16:0) + {0..15} (contiguous).
// C/D (16x16 f32): VGPR v -> M = v + 8*(lane>>4), N = lane&15.
// ---------------------------------------------------------------------------
__device__ __forceinline__ v16bf load_a_frag(const bf16* __restrict__ row, int k0, int hl) {
  const bf16* p = row + k0 + hl * 16;
  v8bf lo = *reinterpret_cast<const v8bf*>(p);
  v8bf hi = *reinterpret_cast<const v8bf*>(p + 8);
  v16bf f;
#pragma unroll
  for (int e = 0; e < 8; ++e) { f[e] = lo[e]; f[e + 8] = hi[e]; }
  return f;
}

__device__ __forceinline__ v16bf load_b_frag(const bf16* __restrict__ row, int k0, int hl) {
  const bf16* p = row + k0 + hl * 16;
  v8bf lo = *reinterpret_cast<const v8bf*>(p);
  v8bf hi = *reinterpret_cast<const v8bf*>(p + 8);
  v16bf f;
#pragma unroll
  for (int e = 0; e < 8; ++e) { f[e] = lo[e]; f[e + 8] = hi[e]; }
  return f;
}

// B fragment out of the padded LDS tile (rows are B^T rows for this n-tile)
__device__ __forceinline__ v16bf load_b_lds(const bf16* __restrict__ base, int t,
                                            int r, int hl) {
  const bf16* bp = base + (t * 16 + r) * BSTRIDE + hl * 16;
  v8bf lo = *reinterpret_cast<const v8bf*>(bp);
  v8bf hi = *reinterpret_cast<const v8bf*>(bp + 8);
  v16bf f;
#pragma unroll
  for (int e = 0; e < 8; ++e) { f[e] = lo[e]; f[e + 8] = hi[e]; }
  return f;
}

__device__ __forceinline__ v8f wmma_bf16(v16bf a, v16bf b, v8f c) {
  return __builtin_amdgcn_wmma_f32_16x16x32_bf16(false, a, false, b, (short)0, c,
                                                 false, false);
}

// ---------------------------------------------------------------------------
// Generic GEMM:  C[M,N] = act(A[M,K] * B^T[N,K]^T + bias) (+ residual)
// 256 threads = 8 waves; block tile 128(M) x 64(N); wave tile 16 x 64.
// B tile staged into double-buffered LDS via CDNA5 async loads (4KB/buffer);
// per iteration: s_wait_asynccnt 0 -> barrier -> issue prefetch -> 4 WMMAs.
// ki loop unrolled x2 so the buffer select (ki&1) is compile-time static.
// ---------------------------------------------------------------------------
template <bool RELU, bool BF16OUT, bool F32OUT, bool ADDRES>
__global__ __launch_bounds__(256) void gemm_bt(
    const bf16* __restrict__ A, const bf16* __restrict__ BT,
    const float* __restrict__ bias, const float* __restrict__ res,
    bf16* __restrict__ Cb, float* __restrict__ Cf, int M, int N, int K) {
  __shared__ __align__(16) bf16 Bs[2][64 * BSTRIDE];
  const int tid  = threadIdx.x;
  const int lane = tid & 31;
  const int wave = tid >> 5;
  const int mBase = blockIdx.x * 128 + wave * 16;
  const int nBase = blockIdx.y * 64;
  const int r  = lane & 15;
  const int hl = lane >> 4;

  int arow = mBase + r; if (arow >= M) arow = M - 1;
  const bf16* aRow = A + (size_t)arow * K;

  // cooperative staging: thread -> (B^T row = tid>>2, 8-elem chunk = tid&3)
  int srow = nBase + (tid >> 2); if (srow >= N) srow = N - 1;
  const bf16* gB = BT + (size_t)srow * K + (tid & 3) * 8;
  const int ldst = (tid >> 2) * BSTRIDE + (tid & 3) * 8;

  v8f acc[4] = {};
  const int nk = K >> 5;

#if USE_ASYNC_LDS
  stage16_async(gB, &Bs[0][ldst]);
  v16bf a_cur = load_a_frag(aRow, 0, hl);
#pragma unroll 2
  for (int ki = 0; ki < nk; ++ki) {
    const int cur = ki & 1;
    wait_async0();
    __syncthreads();  // all waves staged step ki; all finished computing ki-1
    v16bf a_use = a_cur;
    if (ki + 1 < nk) {  // prefetch overlaps the WMMAs below
      stage16_async(gB + (ki + 1) * 32, &Bs[cur ^ 1][ldst]);
      a_cur = load_a_frag(aRow, (ki + 1) * 32, hl);
    }
#pragma unroll
    for (int t = 0; t < 4; ++t)
      acc[t] = wmma_bf16(a_use, load_b_lds(&Bs[cur][0], t, r, hl), acc[t]);
  }
#else
  v8bf  breg  = *reinterpret_cast<const v8bf*>(gB);
  v16bf a_cur = load_a_frag(aRow, 0, hl);
#pragma unroll 2
  for (int ki = 0; ki < nk; ++ki) {
    const int cur = ki & 1;
    *reinterpret_cast<v8bf*>(&Bs[cur][ldst]) = breg;
    __syncthreads();
    v16bf a_use = a_cur;
    if (ki + 1 < nk) {
      breg  = *reinterpret_cast<const v8bf*>(gB + (ki + 1) * 32);
      a_cur = load_a_frag(aRow, (ki + 1) * 32, hl);
    }
#pragma unroll
    for (int t = 0; t < 4; ++t)
      acc[t] = wmma_bf16(a_use, load_b_lds(&Bs[cur][0], t, r, hl), acc[t]);
  }
#endif

#pragma unroll
  for (int t = 0; t < 4; ++t) {
#pragma unroll
    for (int v = 0; v < 8; ++v) {
      int mi = mBase + v + 8 * hl;
      int ni = nBase + t * 16 + r;
      if (mi < M && ni < N) {
        float val = acc[t][v];
        if (bias)  val += bias[ni];
        if (RELU)  val = fmaxf(val, 0.0f);
        if (ADDRES) val += res[(size_t)mi * N + ni];
        if (F32OUT)  Cf[(size_t)mi * N + ni] = val;
        if (BF16OUT) Cb[(size_t)mi * N + ni] = (bf16)val;
      }
    }
  }
}

// ---------------------------------------------------------------------------
// qrel[b,h,i,r] = sum_d q[b,i,h*64+d] * enc_proj[r, h*64+d]   (K=64, direct)
// grid: (LQ/128, ceil(NREL/64), NB*H)
// ---------------------------------------------------------------------------
__global__ __launch_bounds__(256) void qrel_gemm(
    const bf16* __restrict__ q, const bf16* __restrict__ encp,
    float* __restrict__ qrel) {
  const int lane = threadIdx.x & 31, wave = threadIdx.x >> 5;
  const int bh = blockIdx.z, b = bh >> 3, h = bh & 7;
  const int mBase = blockIdx.x * 128 + wave * 16;
  const int nBase = blockIdx.y * 64;
  const int r = lane & 15, hl = lane >> 4;

  const bf16* aRow = q + ((size_t)(b * LQ + mBase + r)) * D + h * DIM;
  const bf16* bRow[4];
#pragma unroll
  for (int t = 0; t < 4; ++t) {
    int rr = nBase + t * 16 + r; if (rr >= NREL) rr = NREL - 1;
    bRow[t] = encp + (size_t)rr * D + h * DIM;
  }
  v8f acc[4] = {};
#pragma unroll
  for (int k0 = 0; k0 < DIM; k0 += 32) {
    v16bf af = load_a_frag(aRow, k0, hl);
#pragma unroll
    for (int t = 0; t < 4; ++t)
      acc[t] = wmma_bf16(af, load_b_frag(bRow[t], k0, hl), acc[t]);
  }
#pragma unroll
  for (int t = 0; t < 4; ++t)
#pragma unroll
    for (int v = 0; v < 8; ++v) {
      int mi = mBase + v + 8 * hl;
      int ni = nBase + t * 16 + r;
      if (ni < NREL)
        qrel[((size_t)bh * LQ + mi) * NREL + ni] = acc[t][v];
    }
}

// ---------------------------------------------------------------------------
// Self-attention logits with fused relative-position bias + causal mask.
// logits = (q.ke + qrel[b,h,i,ridx] + bias_ke[b,j,h] + biaskr[ridx,h]) * scale
// grid: (3, 6, 64)
// ---------------------------------------------------------------------------
__global__ __launch_bounds__(256) void scores_self(
    const bf16* __restrict__ q, const bf16* __restrict__ ke,
    const float* __restrict__ qrel, const float* __restrict__ biaske,
    const float* __restrict__ biaskr, const unsigned char* __restrict__ qmask,
    float* __restrict__ logits) {
  const int lane = threadIdx.x & 31, wave = threadIdx.x >> 5;
  const int bh = blockIdx.z, b = bh >> 3, h = bh & 7;
  const int mBase = blockIdx.x * 128 + wave * 16;
  const int nBase = blockIdx.y * 64;
  const int r = lane & 15, hl = lane >> 4;

  const bf16* aRow = q + ((size_t)(b * LQ + mBase + r)) * D + h * DIM;
  const bf16* bRow[4];
#pragma unroll
  for (int t = 0; t < 4; ++t)
    bRow[t] = ke + ((size_t)(b * LQ + nBase + t * 16 + r)) * D + h * DIM;

  v8f acc[4] = {};
#pragma unroll
  for (int k0 = 0; k0 < DIM; k0 += 32) {
    v16bf af = load_a_frag(aRow, k0, hl);
#pragma unroll
    for (int t = 0; t < 4; ++t)
      acc[t] = wmma_bf16(af, load_b_frag(bRow[t], k0, hl), acc[t]);
  }
#pragma unroll
  for (int t = 0; t < 4; ++t)
#pragma unroll
    for (int v = 0; v < 8; ++v) {
      int i = mBase + v + 8 * hl;
      int j = nBase + t * 16 + r;
      int dd = i - j; dd = dd > REL ? REL : (dd < -REL ? -REL : dd);
      int ri = dd + REL;
      float val = acc[t][v]
                + qrel[((size_t)bh * LQ + i) * NREL + ri]
                + biaske[(size_t)(b * LQ + j) * H + h]
                + biaskr[ri * H + h];
      val *= SCALE;
      bool ok = (j <= i) && (qmask[b * LQ + i] != 0) && (qmask[b * LQ + j] != 0);
      logits[((size_t)bh * LQ + i) * LK + j] = ok ? val : NEGV;
    }
}

// ---------------------------------------------------------------------------
// Cross-attention logits: y . k1;  k1[b,h,j,d] = xv[b,j, h*128 + d]
// ---------------------------------------------------------------------------
__global__ __launch_bounds__(256) void scores_cross(
    const bf16* __restrict__ y, const bf16* __restrict__ xv,
    const unsigned char* __restrict__ qmask, const unsigned char* __restrict__ vmask,
    float* __restrict__ logits) {
  const int lane = threadIdx.x & 31, wave = threadIdx.x >> 5;
  const int bh = blockIdx.z, b = bh >> 3, h = bh & 7;
  const int mBase = blockIdx.x * 128 + wave * 16;
  const int nBase = blockIdx.y * 64;
  const int r = lane & 15, hl = lane >> 4;

  const bf16* aRow = y + ((size_t)(b * LQ + mBase + r)) * D + h * DIM;
  const bf16* bRow[4];
#pragma unroll
  for (int t = 0; t < 4; ++t)
    bRow[t] = xv + ((size_t)(b * LK + nBase + t * 16 + r)) * (2 * D) + h * (2 * DIM);

  v8f acc[4] = {};
#pragma unroll
  for (int k0 = 0; k0 < DIM; k0 += 32) {
    v16bf af = load_a_frag(aRow, k0, hl);
#pragma unroll
    for (int t = 0; t < 4; ++t)
      acc[t] = wmma_bf16(af, load_b_frag(bRow[t], k0, hl), acc[t]);
  }
#pragma unroll
  for (int t = 0; t < 4; ++t)
#pragma unroll
    for (int v = 0; v < 8; ++v) {
      int i = mBase + v + 8 * hl;
      int j = nBase + t * 16 + r;
      float val = acc[t][v] * SCALE;
      bool ok = (qmask[b * LQ + i] != 0) && (vmask[b * LK + j] != 0);
      logits[((size_t)bh * LQ + i) * LK + j] = ok ? val : NEGV;
    }
}

// ---------------------------------------------------------------------------
// out[b,i,h*64+d] = res[...] + sum_j att[bh,i,j] * vT[bh,d,j]
// Same async LDS pipeline as gemm_bt (vT tile shared by 8 waves).
// grid: (3, 1, 64)
// ---------------------------------------------------------------------------
__global__ __launch_bounds__(256) void av_gemm(
    const bf16* __restrict__ att, const bf16* __restrict__ vT,
    const float* __restrict__ res, float* __restrict__ out) {
  __shared__ __align__(16) bf16 Bs[2][64 * BSTRIDE];
  const int tid = threadIdx.x, lane = tid & 31, wave = tid >> 5;
  const int bh = blockIdx.z, b = bh >> 3, h = bh & 7;
  const int mBase = blockIdx.x * 128 + wave * 16;
  const int r = lane & 15, hl = lane >> 4;

  const bf16* aRow = att + ((size_t)bh * LQ + mBase + r) * LK;
  const bf16* gB   = vT + ((size_t)bh * DIM + (tid >> 2)) * LK + (tid & 3) * 8;
  const int ldst = (tid >> 2) * BSTRIDE + (tid & 3) * 8;

  v8f acc[4] = {};
  const int nk = LK >> 5;  // 12

#if USE_ASYNC_LDS
  stage16_async(gB, &Bs[0][ldst]);
  v16bf a_cur = load_a_frag(aRow, 0, hl);
#pragma unroll 2
  for (int ki = 0; ki < nk; ++ki) {
    const int cur = ki & 1;
    wait_async0();
    __syncthreads();
    v16bf a_use = a_cur;
    if (ki + 1 < nk) {
      stage16_async(gB + (ki + 1) * 32, &Bs[cur ^ 1][ldst]);
      a_cur = load_a_frag(aRow, (ki + 1) * 32, hl);
    }
#pragma unroll
    for (int t = 0; t < 4; ++t)
      acc[t] = wmma_bf16(a_use, load_b_lds(&Bs[cur][0], t, r, hl), acc[t]);
  }
#else
  v8bf  breg  = *reinterpret_cast<const v8bf*>(gB);
  v16bf a_cur = load_a_frag(aRow, 0, hl);
#pragma unroll 2
  for (int ki = 0; ki < nk; ++ki) {
    const int cur = ki & 1;
    *reinterpret_cast<v8bf*>(&Bs[cur][ldst]) = breg;
    __syncthreads();
    v16bf a_use = a_cur;
    if (ki + 1 < nk) {
      breg  = *reinterpret_cast<const v8bf*>(gB + (ki + 1) * 32);
      a_cur = load_a_frag(aRow, (ki + 1) * 32, hl);
    }
#pragma unroll
    for (int t = 0; t < 4; ++t)
      acc[t] = wmma_bf16(a_use, load_b_lds(&Bs[cur][0], t, r, hl), acc[t]);
  }
#endif

#pragma unroll
  for (int t = 0; t < 4; ++t)
#pragma unroll
    for (int v = 0; v < 8; ++v) {
      int i = mBase + v + 8 * hl;
      int d = t * 16 + r;
      size_t o = ((size_t)(b * LQ + i)) * D + h * DIM + d;
      out[o] = res[o] + acc[t][v];
    }
}

// ---------------------------------------------------------------------------
// Row softmax (cols = 384), f32 in -> bf16 probabilities out. One block/row.
// ---------------------------------------------------------------------------
__global__ __launch_bounds__(128) void softmax_bf16(
    const float* __restrict__ logits, bf16* __restrict__ att) {
  const int row = blockIdx.x;
  const float* x = logits + (size_t)row * LK;
  __shared__ float sd[128];
  const int t = threadIdx.x;

  float vmax = -3.4e38f;
  for (int c = t; c < LK; c += 128) vmax = fmaxf(vmax, x[c]);
  sd[t] = vmax; __syncthreads();
  for (int s = 64; s > 0; s >>= 1) {
    if (t < s) sd[t] = fmaxf(sd[t], sd[t + s]);
    __syncthreads();
  }
  vmax = sd[0]; __syncthreads();

  float sum = 0.0f;
  for (int c = t; c < LK; c += 128) sum += __expf(x[c] - vmax);
  sd[t] = sum; __syncthreads();
  for (int s = 64; s > 0; s >>= 1) {
    if (t < s) sd[t] += sd[t + s];
    __syncthreads();
  }
  float inv = 1.0f / sd[0];
  for (int c = t; c < LK; c += 128)
    att[(size_t)row * LK + c] = (bf16)(__expf(x[c] - vmax) * inv);
}

// ---------------------------------------------------------------------------
// LayerNorm over D=512, f32 in -> bf16 out. One 256-thread block per row.
// ---------------------------------------------------------------------------
__global__ __launch_bounds__(256) void layernorm_bf16k(
    const float* __restrict__ x, const float* __restrict__ g,
    const float* __restrict__ bb, bf16* __restrict__ out) {
  const int row = blockIdx.x;
  const float* xr = x + (size_t)row * D;
  __shared__ float s1[256], s2[256];
  const int t = threadIdx.x;
  float a0 = xr[t], a1 = xr[t + 256];
  s1[t] = a0 + a1;
  s2[t] = a0 * a0 + a1 * a1;
  __syncthreads();
  for (int s = 128; s > 0; s >>= 1) {
    if (t < s) { s1[t] += s1[t + s]; s2[t] += s2[t + s]; }
    __syncthreads();
  }
  float mean = s1[0] * (1.0f / 512.0f);
  float var  = s2[0] * (1.0f / 512.0f) - mean * mean;
  float rstd = rsqrtf(var + 1.0e-3f);
  out[(size_t)row * D + t]       = (bf16)((a0 - mean) * rstd * g[t] + bb[t]);
  out[(size_t)row * D + t + 256] = (bf16)((a1 - mean) * rstd * g[t + 256] + bb[t + 256]);
}

// ---------------------------------------------------------------------------
// Sinusoid table (201 x 512) -> bf16
// ---------------------------------------------------------------------------
__global__ void sinusoid_bf16(bf16* __restrict__ enc) {
  int idx = blockIdx.x * 256 + threadIdx.x;
  if (idx >= NREL * D) return;
  int c = idx % D, p = idx / D;
  float pos = (float)(p - REL);
  int i = (c < D / 2) ? c : c - D / 2;
  float ang = pos * __expf(-9.2103403719761836f * ((float)i / 256.0f));
  enc[idx] = (bf16)((c < D / 2) ? __sinf(ang) : __cosf(ang));
}

// ---------------------------------------------------------------------------
// Weight convert+transpose: W[K,N] f32 -> WT[N,K] bf16.
// Thread handles 8 consecutive k of one output row -> one 16B store.
// ---------------------------------------------------------------------------
__global__ void w_to_bf16_t(const float* __restrict__ W, bf16* __restrict__ WT,
                            int K, int N) {
  size_t idx = (size_t)blockIdx.x * 256 + threadIdx.x;
  if (idx >= (size_t)K * N / 8) return;
  int k8 = (int)(idx % (K / 8));
  int n  = (int)(idx / (K / 8));
  v8bf v;
#pragma unroll
  for (int e = 0; e < 8; ++e)
    v[e] = (bf16)W[(size_t)(k8 * 8 + e) * N + n];
  *reinterpret_cast<v8bf*>(&WT[(size_t)n * K + k8 * 8]) = v;
}

// kvT[b,h,d,j] = kv[(b*LK+j)*D + h*64 + d]; 8 consecutive j -> one 16B store
__global__ void transpose_kv(const bf16* __restrict__ kv, bf16* __restrict__ kvT) {
  size_t idx = (size_t)blockIdx.x * 256 + threadIdx.x;
  if (idx >= (size_t)NB * H * DIM * LK / 8) return;
  int j8 = (int)(idx % (LK / 8));
  int d  = (int)((idx / (LK / 8)) % DIM);
  int h  = (int)((idx / ((size_t)(LK / 8) * DIM)) % H);
  int b  = (int)(idx / ((size_t)(LK / 8) * DIM * H));
  v8bf v;
#pragma unroll
  for (int e = 0; e < 8; ++e)
    v[e] = kv[((size_t)(b * LK + j8 * 8 + e)) * D + h * DIM + d];
  *reinterpret_cast<v8bf*>(
      &kvT[((size_t)((b * H + h) * DIM + d)) * LK + j8 * 8]) = v;
}

// v1T[b,h,d,j] = xv[(b*LK+j)*1024 + h*128 + 64 + d]
__global__ void transpose_v1(const bf16* __restrict__ xv, bf16* __restrict__ v1T) {
  size_t idx = (size_t)blockIdx.x * 256 + threadIdx.x;
  if (idx >= (size_t)NB * H * DIM * LK / 8) return;
  int j8 = (int)(idx % (LK / 8));
  int d  = (int)((idx / (LK / 8)) % DIM);
  int h  = (int)((idx / ((size_t)(LK / 8) * DIM)) % H);
  int b  = (int)(idx / ((size_t)(LK / 8) * DIM * H));
  v8bf v;
#pragma unroll
  for (int e = 0; e < 8; ++e)
    v[e] = xv[((size_t)(b * LK + j8 * 8 + e)) * (2 * D) + h * (2 * DIM) + DIM + d];
  *reinterpret_cast<v8bf*>(
      &v1T[((size_t)((b * H + h) * DIM + d)) * LK + j8 * 8]) = v;
}

// ---------------------------------------------------------------------------
extern "C" void kernel_launch(void* const* d_in, const int* in_sizes, int n_in,
                              void* d_out, int out_size, void* d_ws, size_t ws_size,
                              hipStream_t stream) {
  (void)in_sizes; (void)n_in; (void)out_size; (void)ws_size;
  const float* queries = (const float*)d_in[0];
  const float* values  = (const float*)d_in[1];
  const unsigned char* qmask = (const unsigned char*)d_in[2];
  const unsigned char* vmask = (const unsigned char*)d_in[3];
  const float* P[34];
  for (int i = 0; i < 34; ++i) P[i] = (const float*)d_in[4 + i];
  enum { LN0G = 0, LN0B, B0W, B0B, Q0W, Q0B, KE0W, KE0B, KV0W, KV0B, KR0W, KR0B,
         AB0W, AB0B, AB1W, AB1B, LN1G, LN1B, B1W1, B1B1, B1W2, B1B2,
         LN2G, LN2B, B2W1, B2B1, B2W2, B2B2, LN3G, LN3B, B3W1, B3B1, B3W2, B3B2 };

  char* ws = (char*)d_ws;
  size_t off = 0;
  auto alloc = [&](size_t bytes) -> char* {
    char* p = ws + off;
    off = (off + bytes + 255) & ~(size_t)255;
    return p;
  };
  auto abf = [&](size_t n) { return (bf16*)alloc(n * sizeof(bf16)); };
  auto af32 = [&](size_t n) { return (float*)alloc(n * sizeof(float)); };

  // bf16 transposed weights
  bf16* b0T   = abf((size_t)512 * 2048);
  bf16* q0T   = abf((size_t)2048 * 512);
  bf16* ke0T  = abf((size_t)2048 * 512);
  bf16* kv0T  = abf((size_t)2048 * 512);
  bf16* kr0T  = abf((size_t)512 * 512);
  bf16* ab0T  = abf((size_t)512 * 8);
  bf16* ab1T  = abf((size_t)512 * 8);
  bf16* b1w1T = abf((size_t)512 * 2048);
  bf16* b1w2T = abf((size_t)2048 * 512);
  bf16* b2w1T = abf((size_t)512 * 2048);
  bf16* b2w2T = abf((size_t)2048 * 1024);
  bf16* b3w1T = abf((size_t)512 * 2048);
  bf16* b3w2T = abf((size_t)2048 * 512);
  // bf16 activations
  bf16* enc  = abf((size_t)NREL * D);
  bf16* encp = abf((size_t)NREL * D);
  bf16* xln  = abf((size_t)ROWS * D);
  bf16* Hbuf = abf((size_t)ROWS * HID);   // reused: X1, h1, h2, h3
  bf16* qb   = abf((size_t)ROWS * D);
  bf16* keb  = abf((size_t)ROWS * D);
  bf16* kvb  = abf((size_t)ROWS * D);
  bf16* ybf  = abf((size_t)ROWS * D);
  bf16* kvT  = abf((size_t)NB * H * DIM * LK);
  bf16* v1T  = abf((size_t)NB * H * DIM * LK);
  bf16* xvb  = abf((size_t)ROWS * 2 * D);
  bf16* attb = abf((size_t)NB * H * LQ * LK);
  // f32 scratch
  float* logits = af32((size_t)NB * H * LQ * LK);
  float* qrel   = af32((size_t)NB * H * LQ * NREL);
  float* biaske = af32((size_t)ROWS * H);
  float* biaskr = af32((size_t)NREL * H);
  float* q1     = af32((size_t)ROWS * D);
  float* q2     = af32((size_t)ROWS * D);

  auto cdiv = [](int a, int b) { return (a + b - 1) / b; };
  hipStream_t s = stream;

  // Weights -> bf16 transposed (W[K,N] -> WT[N,K])
  w_to_bf16_t<<<cdiv(512 * 2048 / 8, 256), 256, 0, s>>>(P[B0W],  b0T,   512,  2048);
  w_to_bf16_t<<<cdiv(2048 * 512 / 8, 256), 256, 0, s>>>(P[Q0W],  q0T,   2048, 512);
  w_to_bf16_t<<<cdiv(2048 * 512 / 8, 256), 256, 0, s>>>(P[KE0W], ke0T,  2048, 512);
  w_to_bf16_t<<<cdiv(2048 * 512 / 8, 256), 256, 0, s>>>(P[KV0W], kv0T,  2048, 512);
  w_to_bf16_t<<<cdiv(512 * 512 / 8,  256), 256, 0, s>>>(P[KR0W], kr0T,  512,  512);
  w_to_bf16_t<<<cdiv(512 * 8 / 8,    256), 256, 0, s>>>(P[AB0W], ab0T,  512,  8);
  w_to_bf16_t<<<cdiv(512 * 8 / 8,    256), 256, 0, s>>>(P[AB1W], ab1T,  512,  8);
  w_to_bf16_t<<<cdiv(512 * 2048 / 8, 256), 256, 0, s>>>(P[B1W1], b1w1T, 512,  2048);
  w_to_bf16_t<<<cdiv(2048 * 512 / 8, 256), 256, 0, s>>>(P[B1W2], b1w2T, 2048, 512);
  w_to_bf16_t<<<cdiv(512 * 2048 / 8, 256), 256, 0, s>>>(P[B2W1], b2w1T, 512,  2048);
  w_to_bf16_t<<<cdiv(2048 * 1024 / 8, 256), 256, 0, s>>>(P[B2W2], b2w2T, 2048, 1024);
  w_to_bf16_t<<<cdiv(512 * 2048 / 8, 256), 256, 0, s>>>(P[B3W1], b3w1T, 512,  2048);
  w_to_bf16_t<<<cdiv(2048 * 512 / 8, 256), 256, 0, s>>>(P[B3W2], b3w2T, 2048, 512);

  // Relative-position tables: enc (201x512) -> enc_proj -> bias_kr table
  sinusoid_bf16<<<cdiv(NREL * D, 256), 256, 0, s>>>(enc);
  gemm_bt<false, true, false, false><<<dim3(cdiv(NREL, 128), cdiv(D, 64)), 256, 0, s>>>(
      enc, kr0T, P[KR0B], nullptr, encp, nullptr, NREL, D, D);
  gemm_bt<false, false, true, false><<<dim3(cdiv(NREL, 128), 1), 256, 0, s>>>(
      encp, ab1T, P[AB1B], nullptr, nullptr, biaskr, NREL, H, D);

  // Block 0: x = relu(LN(queries) @ b0_W); q/ke/kv projections
  layernorm_bf16k<<<ROWS, 256, 0, s>>>(queries, P[LN0G], P[LN0B], xln);
  gemm_bt<true, true, false, false><<<dim3(24, 32), 256, 0, s>>>(
      xln, b0T, P[B0B], nullptr, Hbuf, nullptr, ROWS, HID, D);
  gemm_bt<false, true, false, false><<<dim3(24, 8), 256, 0, s>>>(
      Hbuf, q0T, P[Q0B], nullptr, qb, nullptr, ROWS, D, HID);
  gemm_bt<false, true, false, false><<<dim3(24, 8), 256, 0, s>>>(
      Hbuf, ke0T, P[KE0B], nullptr, keb, nullptr, ROWS, D, HID);
  gemm_bt<false, true, false, false><<<dim3(24, 8), 256, 0, s>>>(
      Hbuf, kv0T, P[KV0B], nullptr, kvb, nullptr, ROWS, D, HID);
  gemm_bt<false, false, true, false><<<dim3(24, 1), 256, 0, s>>>(
      keb, ab0T, P[AB0B], nullptr, nullptr, biaske, ROWS, H, D);

  // Self-attention with relative-position bias
  transpose_kv<<<cdiv(NB * H * DIM * LK / 8, 256), 256, 0, s>>>(kvb, kvT);
  qrel_gemm<<<dim3(3, cdiv(NREL, 64), NB * H), 256, 0, s>>>(qb, encp, qrel);
  scores_self<<<dim3(3, 6, NB * H), 256, 0, s>>>(qb, keb, qrel, biaske, biaskr,
                                                 qmask, logits);
  softmax_bf16<<<NB * H * LQ, 128, 0, s>>>(logits, attb);
  av_gemm<<<dim3(3, 1, NB * H), 256, 0, s>>>(attb, kvT, queries, q1);

  // Block 1 FFN -> cross-attention query y
  layernorm_bf16k<<<ROWS, 256, 0, s>>>(q1, P[LN1G], P[LN1B], xln);
  gemm_bt<true, true, false, false><<<dim3(24, 32), 256, 0, s>>>(
      xln, b1w1T, P[B1B1], nullptr, Hbuf, nullptr, ROWS, HID, D);
  gemm_bt<false, true, false, false><<<dim3(24, 8), 256, 0, s>>>(
      Hbuf, b1w2T, P[B1B2], nullptr, ybf, nullptr, ROWS, D, HID);

  // Block 2 FFN on values -> k1 / v1
  layernorm_bf16k<<<ROWS, 256, 0, s>>>(values, P[LN2G], P[LN2B], xln);
  gemm_bt<true, true, false, false><<<dim3(24, 32), 256, 0, s>>>(
      xln, b2w1T, P[B2B1], nullptr, Hbuf, nullptr, ROWS, HID, D);
  gemm_bt<false, true, false, false><<<dim3(24, 16), 256, 0, s>>>(
      Hbuf, b2w2T, P[B2B2], nullptr, xvb, nullptr, ROWS, 2 * D, HID);
  transpose_v1<<<cdiv(NB * H * DIM * LK / 8, 256), 256, 0, s>>>(xvb, v1T);

  // Cross-attention
  scores_cross<<<dim3(3, 6, NB * H), 256, 0, s>>>(ybf, xvb, qmask, vmask, logits);
  softmax_bf16<<<NB * H * LQ, 128, 0, s>>>(logits, attb);
  av_gemm<<<dim3(3, 1, NB * H), 256, 0, s>>>(attb, v1T, q1, q2);

  // Block 3 FFN + final residual -> d_out (f32)
  layernorm_bf16k<<<ROWS, 256, 0, s>>>(q2, P[LN3G], P[LN3B], xln);
  gemm_bt<true, true, false, false><<<dim3(24, 32), 256, 0, s>>>(
      xln, b3w1T, P[B3B1], nullptr, Hbuf, nullptr, ROWS, HID, D);
  gemm_bt<false, false, true, true><<<dim3(24, 8), 256, 0, s>>>(
      Hbuf, b3w2T, P[B3B2], q2, nullptr, (float*)d_out, ROWS, D, HID);
}
